// TransNeXt_1511828488891
// MI455X (gfx1250) — matrix operand order repaired and checked by reading, not compiled, not run on hardware
//
#include <hip/hip_runtime.h>
#include <hip/hip_bf16.h>

// ---------------- problem constants ----------------
#define B_    4
#define H_    64
#define W_    64
#define C_    512
#define NH_   8
#define HD_   64
#define N_    4096            // H*W
#define LL_   9               // 3x3 window
#define SR_   4
#define PH_   16
#define PW_   16
#define PL_   256             // PH*PW
#define CPB_  512

typedef _Float16 half8  __attribute__((ext_vector_type(8)));
typedef _Float16 half16 __attribute__((ext_vector_type(16)));
typedef float    float8 __attribute__((ext_vector_type(8)));

// ---------------- wave helpers (wave32) ----------------
__device__ inline float wsum32(float v) {
    for (int m = 16; m > 0; m >>= 1) v += __shfl_xor(v, m, 32);
    return v;
}
__device__ inline float wmax32(float v) {
    for (int m = 16; m > 0; m >>= 1) v = fmaxf(v, __shfl_xor(v, m, 32));
    return v;
}

// ---------------- fp32 -> fp16 convert ----------------
__global__ __launch_bounds__(256)
void cvt_f32_f16(const float* __restrict__ in, _Float16* __restrict__ out, long n) {
    long i = (long)blockIdx.x * 256 + threadIdx.x;
    if (i < n) out[i] = (_Float16)in[i];
}

// ---------------- generic batched WMMA GEMM ----------------
// C[M x Nc] = A[M x K] (f16, row-major) * B[Nc x K]^T (f16, [out][in]) + bias
// act: 0 = none, 1 = exact GELU
// grid: (Nc/64, M/128, batches); block 256 = 8 waves.
// Each wave computes a 16x64 strip: 1 A fragment reused over 4 B fragments.
// K must be a multiple of 64: the K loop is explicitly ping-ponged over two
// fragment buffers so there is no register rotation between WMMAs and loads
// of one buffer overlap matrix ops on the other.
__global__ __launch_bounds__(256)
void gemm_wmma_f16f32(const _Float16* __restrict__ A, const _Float16* __restrict__ Bm,
                      float* __restrict__ Cm, const float* __restrict__ bias,
                      int M, int Nc, int K, long sA, long sB, long sC, int act)
{
    const int lane = threadIdx.x & 31;
    const int wav  = threadIdx.x >> 5;
    const int row0 = blockIdx.y * 128 + wav * 16;
    const int col0 = blockIdx.x * 64;
    const long zb  = blockIdx.z;

    const _Float16* Ab = A  + zb * sA;
    const _Float16* Bb = Bm + zb * sB;
    float*          Cb = Cm + zb * sC;

    // A fragment: lane<16 holds row M=lane, K 0..15;
    //             lane>=16 holds row M=lane-16, K 16..31.
    const int arow = row0 + (lane & 15);
    const int kbA  = (lane < 16) ? 0 : 16;
    // B fragment: lane<16 holds col N=lane, K 0..15; lane>=16 K 16..31.
    const int bcol = col0 + (lane & 15);
    const int kbB  = (lane < 16) ? 0 : 16;

    const _Float16* aptr = Ab + (long)arow * K + kbA;
    const _Float16* bptr = Bb + (long)bcol * K + kbB;
    const long bts = (long)16 * K;          // B col-tile stride (16 rows of K)

    float8 acc0 = {0.f,0.f,0.f,0.f,0.f,0.f,0.f,0.f};
    float8 acc1 = acc0, acc2 = acc0, acc3 = acc0;

    union AU { half16 v; half8 h[2]; };

    // ---- preload buffer 0 (k = 0) ----
    AU a0;
    a0.h[0] = *(const half8*)(aptr);
    a0.h[1] = *(const half8*)(aptr + 8);
    half16 b00 = *(const half16*)(bptr);
    half16 b01 = *(const half16*)(bptr + bts);
    half16 b02 = *(const half16*)(bptr + 2 * bts);
    half16 b03 = *(const half16*)(bptr + 3 * bts);

    for (int k0 = 0; k0 < K; k0 += 64) {
        // load buffer 1 @ k0+32 (always valid: K % 64 == 0)
        const int ka = k0 + 32;
        AU a1;
        a1.h[0] = *(const half8*)(aptr + ka);
        a1.h[1] = *(const half8*)(aptr + ka + 8);
        half16 b10 = *(const half16*)(bptr + ka);
        half16 b11 = *(const half16*)(bptr + bts + ka);
        half16 b12 = *(const half16*)(bptr + 2 * bts + ka);
        half16 b13 = *(const half16*)(bptr + 3 * bts + ka);

        acc0 = __builtin_amdgcn_wmma_f32_16x16x32_f16(false, a0.v, false, b00,
                                                      (short)0, acc0, false, false);
        acc1 = __builtin_amdgcn_wmma_f32_16x16x32_f16(false, a0.v, false, b01,
                                                      (short)0, acc1, false, false);
        acc2 = __builtin_amdgcn_wmma_f32_16x16x32_f16(false, a0.v, false, b02,
                                                      (short)0, acc2, false, false);
        acc3 = __builtin_amdgcn_wmma_f32_16x16x32_f16(false, a0.v, false, b03,
                                                      (short)0, acc3, false, false);

        // load buffer 0 @ k0+64 (guarded) while buffer-1 WMMAs run
        const int kb = k0 + 64;
        if (kb < K) {
            __builtin_prefetch(aptr + kb + 32, 0, 0);
            __builtin_prefetch(bptr + kb + 32, 0, 0);
            a0.h[0] = *(const half8*)(aptr + kb);
            a0.h[1] = *(const half8*)(aptr + kb + 8);
            b00 = *(const half16*)(bptr + kb);
            b01 = *(const half16*)(bptr + bts + kb);
            b02 = *(const half16*)(bptr + 2 * bts + kb);
            b03 = *(const half16*)(bptr + 3 * bts + kb);
        }

        acc0 = __builtin_amdgcn_wmma_f32_16x16x32_f16(false, a1.v, false, b10,
                                                      (short)0, acc0, false, false);
        acc1 = __builtin_amdgcn_wmma_f32_16x16x32_f16(false, a1.v, false, b11,
                                                      (short)0, acc1, false, false);
        acc2 = __builtin_amdgcn_wmma_f32_16x16x32_f16(false, a1.v, false, b12,
                                                      (short)0, acc2, false, false);
        acc3 = __builtin_amdgcn_wmma_f32_16x16x32_f16(false, a1.v, false, b13,
                                                      (short)0, acc3, false, false);
    }

    // C layout: VGPR j -> row row0 + j + 8*(lane>=16), col (lane&15) of tile
    const int rbase = row0 + ((lane >> 4) << 3);
    const int cl    = lane & 15;
    float8 accs[4] = {acc0, acc1, acc2, acc3};
#pragma unroll
    for (int t = 0; t < 4; ++t) {
        const int col = col0 + t * 16 + cl;
        const float bv = bias ? bias[col] : 0.f;
#pragma unroll
        for (int j = 0; j < 8; ++j) {
            float v = accs[t][j] + bv;
            if (act == 1) v = 0.5f * v * (1.f + erff(v * 0.70710678118f));
            Cb[(long)(rbase + j) * Nc + col] = v;
        }
    }
}

// ---------------- q normalization / scaling ----------------
// one wave per (b,h,n); lane covers d and d+32
__global__ __launch_bounds__(256)
void qs_kernel(const float* __restrict__ Q, const float* __restrict__ qe,
               const float* __restrict__ temp, float* __restrict__ qn,
               _Float16* __restrict__ qs16, float seq_scale)
{
    int wid  = (int)(((long)blockIdx.x * blockDim.x + threadIdx.x) >> 5);
    int lane = threadIdx.x & 31;
    if (wid >= B_ * NH_ * N_) return;
    int n = wid % N_;
    int h = (wid / N_) % NH_;
    int b = wid / (N_ * NH_);

    const float* qr = Q + ((long)(b * N_ + n)) * C_ + h * HD_;
    float q0 = qr[lane], q1 = qr[lane + 32];
    float ss  = wsum32(q0 * q0 + q1 * q1);
    float inv = 1.f / fmaxf(sqrtf(ss), 1e-12f);
    float t     = temp[h];
    float scale = log1pf(expf(t)) * seq_scale;   // softplus(temperature) * log(N+PL)

    float qn0 = q0 * inv, qn1 = q1 * inv;
    long ob = ((long)(b * NH_ + h) * N_ + n) * HD_;
    qn[ob + lane]        = qn0;
    qn[ob + lane + 32]   = qn1;
    qs16[ob + lane]      = (_Float16)((qn0 + qe[h * HD_ + lane])      * scale);
    qs16[ob + lane + 32] = (_Float16)((qn1 + qe[h * HD_ + lane + 32]) * scale);
}

// ---------------- SRxSR avg-pool + LayerNorm -> xp16 ----------------
__global__ __launch_bounds__(256)
void pool_ln_kernel(const float* __restrict__ S, const float* __restrict__ g,
                    const float* __restrict__ be, _Float16* __restrict__ xp16)
{
    __shared__ float red[256];
    int bp = blockIdx.x;              // b*PL + p
    int b = bp / PL_, p = bp % PL_;
    int ph = p / PW_, pw = p % PW_;
    int t = threadIdx.x;

    float v[2];
#pragma unroll
    for (int c2 = 0; c2 < 2; ++c2) {
        int c = t + c2 * 256;
        float acc = 0.f;
        for (int i = 0; i < SR_; ++i)
            for (int j = 0; j < SR_; ++j) {
                int n = (ph * SR_ + i) * W_ + (pw * SR_ + j);
                acc += S[((long)(b * N_ + n)) * C_ + c];
            }
        v[c2] = acc * (1.f / (SR_ * SR_));
    }
    red[t] = v[0] + v[1];
    __syncthreads();
    for (int s = 128; s > 0; s >>= 1) { if (t < s) red[t] += red[t + s]; __syncthreads(); }
    float mean = red[0] / (float)C_;
    __syncthreads();
    red[t] = (v[0] - mean) * (v[0] - mean) + (v[1] - mean) * (v[1] - mean);
    __syncthreads();
    for (int s = 128; s > 0; s >>= 1) { if (t < s) red[t] += red[t + s]; __syncthreads(); }
    float rinv = rsqrtf(red[0] / (float)C_ + 1e-5f);
#pragma unroll
    for (int c2 = 0; c2 < 2; ++c2) {
        int c = t + c2 * 256;
        xp16[(long)bp * C_ + c] = (_Float16)((v[c2] - mean) * rinv * g[c] + be[c]);
    }
}

// ---------------- k_pool norm + v_pool transpose ----------------
__global__ __launch_bounds__(256)
void kpool_kernel(const float* __restrict__ KVP, _Float16* __restrict__ kpn,
                  _Float16* __restrict__ vpT)
{
    int wid  = (int)(((long)blockIdx.x * blockDim.x + threadIdx.x) >> 5);
    int lane = threadIdx.x & 31;
    if (wid >= B_ * NH_ * PL_) return;
    int m = wid % PL_;
    int h = (wid / PL_) % NH_;
    int b = wid / (PL_ * NH_);

    const float* r = KVP + ((long)(b * PL_ + m)) * (2 * C_);
    float k0 = r[h * HD_ + lane],       k1 = r[h * HD_ + lane + 32];
    float v0 = r[C_ + h * HD_ + lane],  v1 = r[C_ + h * HD_ + lane + 32];
    float inv = 1.f / fmaxf(sqrtf(wsum32(k0 * k0 + k1 * k1)), 1e-12f);

    long bh = (long)(b * NH_ + h);
    long kb = (bh * PL_ + m) * HD_;
    kpn[kb + lane]      = (_Float16)(k0 * inv);
    kpn[kb + lane + 32] = (_Float16)(k1 * inv);
    vpT[(bh * HD_ + lane) * PL_ + m]        = (_Float16)v0;
    vpT[(bh * HD_ + lane + 32) * PL_ + m]   = (_Float16)v1;
}

// ---------------- CPB MLP (rel_table -> per-head bias table) ----------------
__global__ __launch_bounds__(256)
void cpb_kernel(const float* __restrict__ tab, const float* __restrict__ w1,
                const float* __restrict__ b1, const float* __restrict__ w2,
                const float* __restrict__ b2, float* __restrict__ out, int T)
{
    int t = blockIdx.x * 256 + threadIdx.x;
    if (t >= T) return;
    float x0 = tab[(long)t * 2], x1 = tab[(long)t * 2 + 1];
    float acc[NH_];
#pragma unroll
    for (int h = 0; h < NH_; ++h) acc[h] = 0.f;
    for (int j = 0; j < CPB_; ++j) {
        float hv = fmaxf(0.f, x0 * w1[j * 2] + x1 * w1[j * 2 + 1] + b1[j]);
#pragma unroll
        for (int h = 0; h < NH_; ++h) acc[h] += hv * w2[h * CPB_ + j];
    }
#pragma unroll
    for (int h = 0; h < NH_; ++h) out[(long)h * T + t] = acc[h] + b2[h];
}

// ---------------- local 3x3 attention scores ----------------
__global__ __launch_bounds__(256)
void attn_local_kernel(const _Float16* __restrict__ qs16, const float* __restrict__ KV,
                       const float* __restrict__ rbl, float* __restrict__ aL)
{
    int wid  = (int)(((long)blockIdx.x * blockDim.x + threadIdx.x) >> 5);
    int lane = threadIdx.x & 31;
    if (wid >= B_ * NH_ * N_) return;
    int n = wid % N_;
    int h = (wid / N_) % NH_;
    int b = wid / (N_ * NH_);
    int i = n / W_, j = n % W_;

    long qb = ((long)(b * NH_ + h) * N_ + n) * HD_;
    float q0 = (float)qs16[qb + lane], q1 = (float)qs16[qb + lane + 32];
    long ob = ((long)(b * NH_ + h) * N_ + n) * LL_;

#pragma unroll
    for (int kk = 0; kk < LL_; ++kk) {
        int ii = i + kk / 3 - 1, jj = j + kk % 3 - 1;
        float val;
        if (ii < 0 || ii >= H_ || jj < 0 || jj >= W_) {
            val = -__builtin_inff();
        } else {
            const float* kr = KV + ((long)(b * N_ + ii * W_ + jj)) * (2 * C_) + h * HD_;
            float k0 = kr[lane], k1 = kr[lane + 32];
            float ss = wsum32(k0 * k0 + k1 * k1);
            float dt = wsum32(q0 * k0 + q1 * k1);
            val = dt / fmaxf(sqrtf(ss), 1e-12f) + rbl[h * LL_ + kk];
        }
        if (lane == 0) aL[ob + kk] = val;
    }
}

// ---------------- fused 265-way softmax + learnable tokens + local V agg ----
__global__ __launch_bounds__(256)
void softmax_combine_kernel(const float* __restrict__ aP, const float* __restrict__ aL,
                            const float* __restrict__ cpb, const int* __restrict__ ridx,
                            const float* __restrict__ qn, const float* __restrict__ lt,
                            const float* __restrict__ lb, const float* __restrict__ KV,
                            _Float16* __restrict__ ap16, float* __restrict__ y32, int T)
{
    int wid  = (int)(((long)blockIdx.x * blockDim.x + threadIdx.x) >> 5);
    int lane = threadIdx.x & 31;
    if (wid >= B_ * NH_ * N_) return;
    int n = wid % N_;
    int h = (wid / N_) % NH_;
    int b = wid / (N_ * NH_);

    long rowP = ((long)(b * NH_ + h) * N_ + n) * PL_;
    const int*   rix = ridx + (long)n * PL_;
    const float* cr  = cpb + (long)h * T;

    float s[8];
#pragma unroll
    for (int t2 = 0; t2 < 8; ++t2) {
        int m = lane + t2 * 32;
        s[t2] = aP[rowP + m] + cr[rix[m]];
    }
    float lv = (lane < LL_) ? aL[((long)(b * NH_ + h) * N_ + n) * LL_ + lane]
                            : -__builtin_inff();
    float mx = lv;
#pragma unroll
    for (int t2 = 0; t2 < 8; ++t2) mx = fmaxf(mx, s[t2]);
    mx = wmax32(mx);

    float le  = (lane < LL_) ? __expf(lv - mx) : 0.f;
    float tot = le;
    float e[8];
#pragma unroll
    for (int t2 = 0; t2 < 8; ++t2) { e[t2] = __expf(s[t2] - mx); tot += e[t2]; }
    tot = wsum32(tot);
    float inv = 1.f / tot;

#pragma unroll
    for (int t2 = 0; t2 < 8; ++t2)
        ap16[rowP + lane + t2 * 32] = (_Float16)(e[t2] * inv);

    // a_l = softmax_local + q_norm . learnable_tokens + learnable_bias
    long qb = ((long)(b * NH_ + h) * N_ + n) * HD_;
    float al = 0.f;
    if (lane < LL_) {
        float d = 0.f;
        for (int dd = 0; dd < HD_; ++dd)
            d += qn[qb + dd] * lt[(h * HD_ + dd) * LL_ + lane];
        al = le * inv + d + lb[h * LL_ + lane];
    }
    float alk[LL_];
#pragma unroll
    for (int kk = 0; kk < LL_; ++kk) alk[kk] = __shfl(al, kk, 32);

    int i = n / W_, jx = n % W_;
    float x0 = 0.f, x1 = 0.f;
#pragma unroll
    for (int kk = 0; kk < LL_; ++kk) {
        int ii = i + kk / 3 - 1, jj = jx + kk % 3 - 1;
        if (ii < 0 || ii >= H_ || jj < 0 || jj >= W_) continue;  // padded v == 0
        const float* vr = KV + ((long)(b * N_ + ii * W_ + jj)) * (2 * C_) + C_ + h * HD_;
        x0 += alk[kk] * vr[lane];
        x1 += alk[kk] * vr[lane + 32];
    }
    long yb = ((long)(b * N_ + n)) * C_ + h * HD_;
    y32[yb + lane]      = x0;
    y32[yb + lane + 32] = x1;
}

// ---------------- x_l + x_p combine -> f16 ----------------
__global__ __launch_bounds__(256)
void combine_kernel(const float* __restrict__ y32, const float* __restrict__ xpO,
                    _Float16* __restrict__ y16)
{
    long i = (long)blockIdx.x * 256 + threadIdx.x;
    long total = (long)B_ * N_ * C_;
    if (i >= total) return;
    int c = (int)(i % C_);
    long r = i / C_;
    int n = (int)(r % N_);
    int b = (int)(r / N_);
    int h = c / HD_, d = c % HD_;
    y16[i] = (_Float16)(y32[i] + xpO[(((long)(b * NH_ + h) * N_) + n) * HD_ + d]);
}

// ===================================================================
extern "C" void kernel_launch(void* const* d_in, const int* in_sizes, int n_in,
                              void* d_out, int out_size, void* d_ws, size_t ws_size,
                              hipStream_t stream) {
    // inputs (setup_inputs order)
    const float* x      = (const float*)d_in[0];
    const float* q_w    = (const float*)d_in[1];
    const float* q_b    = (const float*)d_in[2];
    const float* kv_w   = (const float*)d_in[3];
    const float* kv_b   = (const float*)d_in[4];
    const float* temp   = (const float*)d_in[5];
    const float* qe     = (const float*)d_in[6];
    const float* rbl    = (const float*)d_in[7];
    const float* cpb1_w = (const float*)d_in[8];
    const float* cpb1_b = (const float*)d_in[9];
    const float* cpb2_w = (const float*)d_in[10];
    const float* cpb2_b = (const float*)d_in[11];
    const float* sr_w   = (const float*)d_in[12];
    const float* sr_b   = (const float*)d_in[13];
    const float* ln_g   = (const float*)d_in[14];
    const float* ln_b   = (const float*)d_in[15];
    const float* lt     = (const float*)d_in[16];
    const float* lb     = (const float*)d_in[17];
    const float* proj_w = (const float*)d_in[18];
    const float* proj_b = (const float*)d_in[19];
    const float* rtab   = (const float*)d_in[20];
    const int*   ridx   = (const int*)d_in[21];
    // d_in[22] = local_mask (recomputed analytically in-kernel)

    const int T = in_sizes[20] / 2;      // rows of rel_table
    float* out = (float*)d_out;

    // ---- workspace carve ----
    size_t off = 0;
    auto carve = [&](size_t bytes) -> char* {
        off = (off + 255) & ~(size_t)255;
        char* p = (char*)d_ws + off;
        off += bytes;
        return p;
    };
    _Float16* x16   = (_Float16*)carve((size_t)B_ * N_ * C_ * 2);
    _Float16* qw16  = (_Float16*)carve((size_t)C_ * C_ * 2);
    _Float16* kvw16 = (_Float16*)carve((size_t)2 * C_ * C_ * 2);
    _Float16* srw16 = (_Float16*)carve((size_t)C_ * C_ * 2);
    _Float16* pjw16 = (_Float16*)carve((size_t)C_ * C_ * 2);
    float*    Qf    = (float*)carve((size_t)B_ * N_ * C_ * 4);
    float*    KVf   = (float*)carve((size_t)B_ * N_ * 2 * C_ * 4);
    float*    SRf   = (float*)carve((size_t)B_ * N_ * C_ * 4);
    float*    qn32  = (float*)carve((size_t)B_ * NH_ * N_ * HD_ * 4);
    _Float16* qs16  = (_Float16*)carve((size_t)B_ * NH_ * N_ * HD_ * 2);
    _Float16* xp16  = (_Float16*)carve((size_t)B_ * PL_ * C_ * 2);
    float*    KVPf  = (float*)carve((size_t)B_ * PL_ * 2 * C_ * 4);
    _Float16* kpn16 = (_Float16*)carve((size_t)B_ * NH_ * PL_ * HD_ * 2);
    _Float16* vpT16 = (_Float16*)carve((size_t)B_ * NH_ * HD_ * PL_ * 2);
    float*    cpbT  = (float*)carve((size_t)NH_ * T * 4);
    float*    aL    = (float*)carve((size_t)B_ * NH_ * N_ * LL_ * 4);
    float*    aP    = (float*)carve((size_t)B_ * NH_ * N_ * PL_ * 4);
    _Float16* ap16  = (_Float16*)carve((size_t)B_ * NH_ * N_ * PL_ * 2);
    float*    y32   = (float*)carve((size_t)B_ * N_ * C_ * 4);
    float*    xpO   = (float*)carve((size_t)B_ * NH_ * N_ * HD_ * 4);
    _Float16* y16   = (_Float16*)carve((size_t)B_ * N_ * C_ * 2);
    (void)ws_size;

    const float seq_scale = logf((float)(N_ + PL_));

    // ---- 1. fp16 converts ----
    {
        long n;
        n = (long)B_ * N_ * C_;
        cvt_f32_f16<<<(int)((n + 255) / 256), 256, 0, stream>>>(x, x16, n);
        n = (long)C_ * C_;
        cvt_f32_f16<<<(int)((n + 255) / 256), 256, 0, stream>>>(q_w, qw16, n);
        n = (long)2 * C_ * C_;
        cvt_f32_f16<<<(int)((n + 255) / 256), 256, 0, stream>>>(kv_w, kvw16, n);
        n = (long)C_ * C_;
        cvt_f32_f16<<<(int)((n + 255) / 256), 256, 0, stream>>>(sr_w, srw16, n);
        n = (long)C_ * C_;
        cvt_f32_f16<<<(int)((n + 255) / 256), 256, 0, stream>>>(proj_w, pjw16, n);
    }

    const int M = B_ * N_;     // 16384
    // ---- 2. Q = x @ q_w^T + q_b ----
    gemm_wmma_f16f32<<<dim3(C_ / 64, M / 128, 1), 256, 0, stream>>>(
        x16, qw16, Qf, q_b, M, C_, C_, 0, 0, 0, 0);
    // ---- 3. KV = x @ kv_w^T + kv_b ----
    gemm_wmma_f16f32<<<dim3(2 * C_ / 64, M / 128, 1), 256, 0, stream>>>(
        x16, kvw16, KVf, kv_b, M, 2 * C_, C_, 0, 0, 0, 0);
    // ---- 4. SR = gelu(x @ sr_w^T + sr_b) ----
    gemm_wmma_f16f32<<<dim3(C_ / 64, M / 128, 1), 256, 0, stream>>>(
        x16, srw16, SRf, sr_b, M, C_, C_, 0, 0, 0, 1);

    // ---- 5. q_norm / q_s ----
    qs_kernel<<<(B_ * NH_ * N_ * 32) / 256, 256, 0, stream>>>(
        Qf, qe, temp, qn32, qs16, seq_scale);

    // ---- 6. pool + layernorm -> xp16 ----
    pool_ln_kernel<<<B_ * PL_, 256, 0, stream>>>(SRf, ln_g, ln_b, xp16);

    // ---- 7. KVP = xp @ kv_w^T + kv_b ----
    gemm_wmma_f16f32<<<dim3(2 * C_ / 64, (B_ * PL_) / 128, 1), 256, 0, stream>>>(
        xp16, kvw16, KVPf, kv_b, B_ * PL_, 2 * C_, C_, 0, 0, 0, 0);

    // ---- 8. k_pool norm + v_pool^T ----
    kpool_kernel<<<(B_ * NH_ * PL_ * 32) / 256, 256, 0, stream>>>(KVPf, kpn16, vpT16);

    // ---- 9. CPB bias table ----
    cpb_kernel<<<(T + 255) / 256, 256, 0, stream>>>(
        rtab, cpb1_w, cpb1_b, cpb2_w, cpb2_b, cpbT, T);

    // ---- 10. local attention scores ----
    attn_local_kernel<<<(B_ * NH_ * N_ * 32) / 256, 256, 0, stream>>>(
        qs16, KVf, rbl, aL);

    // ---- 11. attn_pool = q_s @ k_pool_norm^T (batched over B*NH) ----
    gemm_wmma_f16f32<<<dim3(PL_ / 64, N_ / 128, B_ * NH_), 256, 0, stream>>>(
        qs16, kpn16, aP, nullptr, N_, PL_, HD_,
        (long)N_ * HD_, (long)PL_ * HD_, (long)N_ * PL_, 0);

    // ---- 12. softmax(265) + learnable tokens + local V aggregation ----
    softmax_combine_kernel<<<(B_ * NH_ * N_ * 32) / 256, 256, 0, stream>>>(
        aP, aL, cpbT, ridx, qn32, lt, lb, KVf, ap16, y32, T);

    // ---- 13. x_p = a_p @ v_pool (batched over B*NH) ----
    gemm_wmma_f16f32<<<dim3(HD_ / 64, N_ / 128, B_ * NH_), 256, 0, stream>>>(
        ap16, vpT16, xpO, nullptr, N_, HD_, PL_,
        (long)N_ * PL_, (long)HD_ * PL_, (long)N_ * HD_, 0);

    // ---- 14. y = x_l + x_p -> f16 ----
    {
        long n = (long)B_ * N_ * C_;
        combine_kernel<<<(int)((n + 255) / 256), 256, 0, stream>>>(y32, xpO, y16);
    }

    // ---- 15. out = y @ proj_w^T + proj_b ----
    gemm_wmma_f16f32<<<dim3(C_ / 64, M / 128, 1), 256, 0, stream>>>(
        y16, pjw16, out, proj_b, M, C_, C_, 0, 0, 0, 0);
}
